// MugsQueue_48670569398436
// MI455X (gfx1250) — compile-verified, matrix operand order checked by reading, not run on hardware
//
#include <hip/hip_runtime.h>
#include <hip/hip_bf16.h>

// ---------------------------------------------------------------------------
// MugsQueue on MI455X (gfx1250, wave32, WMMA bf16)
//   SIZE=65536, DIM=256, TOPK=8, N=4096, BATCH=2048
// Outputs (concat): neighbors[4096*8*256] f32 | new_queue[65536*256] f32 | new_ptr (i32 bits)
// Workspace: nxb bf16[4096*256] (2MB) | nqb bf16[65536*256] (32MB)
// ---------------------------------------------------------------------------

#define MQ_SIZE  65536
#define MQ_DIM   256
#define MQ_TOPK  8
#define MQ_N     4096
#define MQ_CPAD  20      // staged C row stride (floats): conflict-free + 16B aligned

typedef __attribute__((ext_vector_type(16))) __bf16 v16bf;
typedef __attribute__((ext_vector_type(8)))  float  v8f;

union FragU {
    v16bf v;
    uint4 q[2];
};

__device__ __forceinline__ v16bf mq_ld_frag(const unsigned short* p0,
                                            const unsigned short* p1) {
    FragU f;
    f.q[0] = *reinterpret_cast<const uint4*>(p0);   // elems 0-7  (K = khalf*8 .. +7)
    f.q[1] = *reinterpret_cast<const uint4*>(p1);   // elems 8-15 (K = khalf*8+16 .. +23)
    return f.v;
}

__device__ __forceinline__ unsigned mq_bfpack(float a, float b) {
    unsigned ua = __float_as_uint(a);
    unsigned ub = __float_as_uint(b);
    ua = (ua + 0x7FFFu + ((ua >> 16) & 1u)) >> 16;   // RNE
    ub = (ub + 0x7FFFu + ((ub >> 16) & 1u)) >> 16;
    return ua | (ub << 16);
}

__device__ __forceinline__ void mq_topk_insert(float (&tv)[8], int (&ti)[8],
                                               float v, int gi) {
    if (v > tv[7]) {
        tv[7] = v; ti[7] = gi;
#pragma unroll
        for (int j = 7; j > 0; --j) {
            if (tv[j] > tv[j - 1]) {
                float a = tv[j]; tv[j] = tv[j - 1]; tv[j - 1] = a;
                int   b = ti[j]; ti[j] = ti[j - 1]; ti[j - 1] = b;
            }
        }
    }
}

// ---------------------------------------------------------------------------
// Kernel 1: L2-normalize rows of x and queue, pack to bf16. One wave per row.
// ---------------------------------------------------------------------------
__global__ __launch_bounds__(256)
void mq_norm_bf16(const float* __restrict__ x, const float* __restrict__ queue,
                  unsigned short* __restrict__ nxb, unsigned short* __restrict__ nqb) {
    const int row  = blockIdx.x * 8 + (threadIdx.x >> 5);
    const int lane = threadIdx.x & 31;

    const float* src;
    unsigned short* dst;
    if (row < MQ_N) {
        src = x   + (size_t)row * MQ_DIM;
        dst = nxb + (size_t)row * MQ_DIM;
    } else {
        const size_t qr = (size_t)(row - MQ_N);
        src = queue + qr * MQ_DIM;
        dst = nqb   + qr * MQ_DIM;
    }

    float4 a = *reinterpret_cast<const float4*>(src + lane * 8);
    float4 b = *reinterpret_cast<const float4*>(src + lane * 8 + 4);
    float ss = a.x*a.x + a.y*a.y + a.z*a.z + a.w*a.w
             + b.x*b.x + b.y*b.y + b.z*b.z + b.w*b.w;
#pragma unroll
    for (int off = 16; off > 0; off >>= 1) ss += __shfl_xor(ss, off, 32);

    const float s = 1.0f / fmaxf(sqrtf(ss), 1e-12f);
    uint4 o;
    o.x = mq_bfpack(a.x * s, a.y * s);
    o.y = mq_bfpack(a.z * s, a.w * s);
    o.z = mq_bfpack(b.x * s, b.y * s);
    o.w = mq_bfpack(b.z * s, b.w * s);
    *reinterpret_cast<uint4*>(dst + lane * 8) = o;
}

// ---------------------------------------------------------------------------
// Kernel 2: fused sim = nx @ nq^T (WMMA bf16) + per-row top-8 + neighbor gather.
// 256 blocks x 256 threads (8 waves). Block owns 16 x-rows; waves stride N-tiles.
// All 8 B fragments use distinct registers so all 16 global_load_b128 pipeline
// ahead of the WMMA chain.
// ---------------------------------------------------------------------------
__global__ __launch_bounds__(256)
void mq_simtopk(const unsigned short* __restrict__ nxb,
                const unsigned short* __restrict__ nqb,
                const float* __restrict__ queue,
                float* __restrict__ out_nb) {
    __shared__ unsigned short sA[16 * MQ_DIM];   // 8 KB  : bf16 A panel (16 x-rows)
    __shared__ float sC[8][16 * MQ_CPAD];        // 10 KB : per-wave staged C tile
    __shared__ float sPV[16 * 16 * 8];           // 8 KB  : partial top-8 values
    __shared__ int   sPI[16 * 16 * 8];           // 8 KB  : partial top-8 indices
    __shared__ int   sTop[16 * 8];               // final indices

    const int tid   = threadIdx.x;
    const int wave  = tid >> 5;
    const int lane  = tid & 31;
    const int r     = lane & 15;     // row (A) / column (B) within tile
    const int half  = lane >> 4;     // K-half selector per ISA 16-bit layout
    const int mBase = blockIdx.x * 16;

    // Stage A panel (16 x 256 bf16) into LDS.
    {
        const uint4* src = reinterpret_cast<const uint4*>(nxb + (size_t)mBase * MQ_DIM);
        uint4* dst = reinterpret_cast<uint4*>(sA);
        for (int i = tid; i < 16 * MQ_DIM / 8; i += 256) dst[i] = src[i];
    }
    __syncthreads();

    // Preload all 8 A fragments (K = 256 -> 8 k-steps of 32) into VGPRs.
    v16bf afrag[8];
#pragma unroll
    for (int kk = 0; kk < 8; ++kk) {
        const unsigned short* pa = &sA[r * MQ_DIM + kk * 32 + half * 8];
        afrag[kk] = mq_ld_frag(pa, pa + 16);
    }

    float tv[8]; int ti[8];
#pragma unroll
    for (int j = 0; j < 8; ++j) { tv[j] = -1e30f; ti[j] = 0; }

    float* myC = &sC[wave][0];

    for (int nt = wave; nt < MQ_SIZE / 16; nt += 8) {
        const int colBase = nt << 4;
        const unsigned short* pb =
            nqb + (size_t)(colBase + r) * MQ_DIM + half * 8;

        if (nt + 8 < MQ_SIZE / 16) {
            // Prefetch next B tile for this wave -> global_prefetch_b8
            __builtin_prefetch(nqb + (size_t)(colBase + 128 + r) * MQ_DIM, 0, 1);
        }

        // Issue ALL B-fragment loads first (distinct regs -> 16 loads in flight),
        // then drain through the 8-deep WMMA accumulation chain.
        v16bf bfrag[8];
#pragma unroll
        for (int kk = 0; kk < 8; ++kk)
            bfrag[kk] = mq_ld_frag(pb + kk * 32, pb + kk * 32 + 16);

        v8f c;
#pragma unroll
        for (int i = 0; i < 8; ++i) c[i] = 0.0f;

#pragma unroll
        for (int kk = 0; kk < 8; ++kk) {
            c = __builtin_amdgcn_wmma_f32_16x16x32_bf16(
                    false, afrag[kk], false, bfrag[kk],
                    (short)0, c, false, false);
        }

        // Stage C tile: VGPR i holds M = i + 8*half (lanes 0-15 / 16-31), N = r.
#pragma unroll
        for (int i = 0; i < 8; ++i) myC[(i + half * 8) * MQ_CPAD + r] = c[i];

        // Scan (same wave; LDS ops in-order per wave): lane covers row r,
        // columns [half*8, half*8+8). Two ds_load_b128 instead of 8 b32.
        {
            const float4* rowp =
                reinterpret_cast<const float4*>(&myC[r * MQ_CPAD + half * 8]);
            const float4 q0 = rowp[0];
            const float4 q1 = rowp[1];
            const int cb = colBase + half * 8;
            mq_topk_insert(tv, ti, q0.x, cb + 0);
            mq_topk_insert(tv, ti, q0.y, cb + 1);
            mq_topk_insert(tv, ti, q0.z, cb + 2);
            mq_topk_insert(tv, ti, q0.w, cb + 3);
            mq_topk_insert(tv, ti, q1.x, cb + 4);
            mq_topk_insert(tv, ti, q1.y, cb + 5);
            mq_topk_insert(tv, ti, q1.z, cb + 6);
            mq_topk_insert(tv, ti, q1.w, cb + 7);
        }
    }

    // Dump partial lists: 16 lists per row (8 waves x 2 halves).
    {
        const int list = wave * 2 + half;
#pragma unroll
        for (int j = 0; j < 8; ++j) {
            sPV[(r * 16 + list) * 8 + j] = tv[j];
            sPI[(r * 16 + list) * 8 + j] = ti[j];
        }
    }
    __syncthreads();

    // Final merge: thread t (<16) reduces 128 candidates for its row.
    if (tid < 16) {
        float fv[8]; int fi[8];
#pragma unroll
        for (int j = 0; j < 8; ++j) { fv[j] = -1e30f; fi[j] = 0; }
        for (int k2 = 0; k2 < 128; ++k2)
            mq_topk_insert(fv, fi, sPV[tid * 128 + k2], sPI[tid * 128 + k2]);
#pragma unroll
        for (int j = 0; j < 8; ++j) sTop[tid * 8 + j] = fi[j];
    }
    __syncthreads();

    // Gather neighbors from the ORIGINAL fp32 queue: 16 rows x 8 x 256 floats.
    const float4* qv = reinterpret_cast<const float4*>(queue);
    float4* ov = reinterpret_cast<float4*>(out_nb) + (size_t)mBase * 512;
    for (int e = tid; e < 16 * 8 * 64; e += 256) {
        const int row = e >> 9;          // /512
        const int rem = e & 511;
        const int k   = rem >> 6;
        const int d4  = rem & 63;
        const int qrow = sTop[row * 8 + k];
        ov[e] = qv[(size_t)qrow * 64 + d4];
    }
}

// ---------------------------------------------------------------------------
// Kernel 3: new_queue = ring-overwrite copy; new_ptr. One float4 per thread.
// ---------------------------------------------------------------------------
__global__ __launch_bounds__(256)
void mq_update(const float* __restrict__ x, const float* __restrict__ queue,
               const int* __restrict__ ptr, const int* __restrict__ pbatch,
               float* __restrict__ out_q, int* __restrict__ out_p) {
    const int t   = blockIdx.x * blockDim.x + threadIdx.x;  // float4 granularity
    const int row = t >> 6;
    const int c4  = t & 63;
    const int p     = ptr[0];
    const int batch = pbatch[0];

    int d = row - p;
    if (d < 0) d += MQ_SIZE;                       // (row - ptr) mod SIZE

    const float4* src = (d < batch)
        ? reinterpret_cast<const float4*>(x     + (size_t)d   * MQ_DIM)
        : reinterpret_cast<const float4*>(queue + (size_t)row * MQ_DIM);
    reinterpret_cast<float4*>(out_q + (size_t)row * MQ_DIM)[c4] = src[c4];

    if (t == 0) *out_p = (p + batch) & (MQ_SIZE - 1);
}

// ---------------------------------------------------------------------------
extern "C" void kernel_launch(void* const* d_in, const int* in_sizes, int n_in,
                              void* d_out, int out_size, void* d_ws, size_t ws_size,
                              hipStream_t stream) {
    (void)in_sizes; (void)n_in; (void)out_size; (void)ws_size;

    const float* x      = (const float*)d_in[0];
    const float* queue  = (const float*)d_in[1];
    const int*   ptr    = (const int*)d_in[2];
    const int*   pbatch = (const int*)d_in[3];

    unsigned short* nxb = (unsigned short*)d_ws;                    //  2 MB
    unsigned short* nqb = nxb + (size_t)MQ_N * MQ_DIM;              // 32 MB

    float* out_nb = (float*)d_out;                                  // 4096*8*256
    float* out_q  = out_nb + (size_t)MQ_N * MQ_TOPK * MQ_DIM;       // 65536*256
    int*   out_p  = (int*)(out_q + (size_t)MQ_SIZE * MQ_DIM);       // 1

    // 1) normalize + bf16 pack: (4096 + 65536) rows, 8 rows (waves) per block
    mq_norm_bf16<<<(MQ_N + MQ_SIZE) / 8, 256, 0, stream>>>(x, queue, nxb, nqb);

    // 2) fused WMMA similarity + top-8 + neighbor gather
    mq_simtopk<<<MQ_N / 16, 256, 0, stream>>>(nxb, nqb, queue, out_nb);

    // 3) ring-buffer queue update + new ptr
    mq_update<<<(MQ_SIZE * (MQ_DIM / 4)) / 256, 256, 0, stream>>>(
        x, queue, ptr, pbatch, out_q, out_p);
}
